// ExhaustiveModel_53223234732240
// MI455X (gfx1250) — matrix operand order repaired and checked by reading, not compiled, not run on hardware
//
#include <hip/hip_runtime.h>
#include <hip/hip_bf16.h>

typedef __attribute__((ext_vector_type(16))) _Float16 v16h;
typedef __attribute__((ext_vector_type(8)))  float    v8f;

#define BB    16
#define SS    128
#define TWC   16
#define EDIM  300
#define CEDIM 50
#define CHH   50
#define HID   256
#define G4    1024      // 4*HID
#define DIN   400       // EDIM + 2*CHH
#define KPAD  416       // DIN padded to multiple of 32
#define H2    512
#define NTAGS 10
#define NWIN  1928
#define CLFK  1536
#define KT_IH (KPAD / 32)   // 13 k-tiles, input projection
#define KT_HH (HID / 32)    // 8  k-tiles, recurrence
#define KT_CL (CLFK / 32)   // 48 k-tiles, classifier
#define NT_G  (G4 / 16)     // 64 n-tiles over the gate dimension

// ---------------------------------------------------------------- math helpers
__device__ __forceinline__ float sigf(float x) { return 1.0f / (1.0f + __expf(-x)); }
__device__ __forceinline__ float tanh_(float x) {
    x = fminf(fmaxf(x, -15.f), 15.f);
    float e = __expf(2.f * x);
    return (e - 1.f) / (e + 1.f);
}

// ------------------------------------------------------- WMMA fragment helpers
// B fragments are pre-packed: packed[(tile*32 + lane)*16 + j] = the 16 f16
// values lane `lane` feeds into the WMMA for that 32x16 tile. One contiguous
// 32-byte load per lane (2 x global_load_b128).
__device__ __forceinline__ v16h load_b_packed(const _Float16* packed, int tile, int lane) {
    return *((const v16h*)packed + (size_t)tile * 32 + lane);
}
__device__ __forceinline__ v8f wmma16(v16h a, v16h b, v8f c) {
    return __builtin_amdgcn_wmma_f32_16x16x32_f16(false, a, false, b, (short)0, c, false, false);
}

// -------------------------- weight pack: convert f32 -> f16 fragment layout
// Source weights are [N, K] (out-major); B matrix is B[k][n] = W[n][k].
// Packed layout per tile (nt,kt): lane = within/16, j = within%16,
//   k = kt*32 + (lane>=16 ? 8 : 0) + (j&7) + (j>=8 ? 16 : 0),  n = nt*16 + (lane&15).
__global__ void k_pack_weights(const float* wihf, const float* wihb,
                               const float* whhf, const float* whhb,
                               const float* clfw,
                               _Float16* wihP, _Float16* whhP, _Float16* clfP) {
    const int N1 = 2 * KPAD * G4;      // wihP (both dirs)
    const int N2 = 2 * HID * G4;       // whhP (both dirs)
    const int N3 = CLFK * 16;          // clfP
    int total = N1 + N2 + N3;
    for (int i = blockIdx.x * blockDim.x + threadIdx.x; i < total;
         i += gridDim.x * blockDim.x) {
        if (i < N1) {
            int d = i / (KPAD * G4), t = i % (KPAD * G4);
            int tile = t / 512, within = t % 512;
            int lane = within / 16, j = within % 16;
            int kt = tile % KT_IH, nt = tile / KT_IH;
            int k = kt * 32 + ((lane >> 4) << 3) + (j & 7) + ((j >> 3) << 4);
            int n = nt * 16 + (lane & 15);
            const float* w = d ? wihb : wihf;              // [G4, DIN]
            wihP[i] = (k < DIN) ? (_Float16)w[(size_t)n * DIN + k] : (_Float16)0.f;
        } else if (i < N1 + N2) {
            int jj = i - N1;
            int d = jj / (HID * G4), t = jj % (HID * G4);
            int tile = t / 512, within = t % 512;
            int lane = within / 16, j = within % 16;
            int kt = tile % KT_HH, nt = tile / KT_HH;
            int k = kt * 32 + ((lane >> 4) << 3) + (j & 7) + ((j >> 3) << 4);
            int n = nt * 16 + (lane & 15);
            const float* w = d ? whhb : whhf;              // [G4, HID]
            whhP[jj] = (_Float16)w[(size_t)n * HID + k];
        } else {
            int jj = i - N1 - N2;
            int tile = jj / 512, within = jj % 512;
            int lane = within / 16, j = within % 16;
            int kt = tile;                                 // NT = 1
            int k = kt * 32 + ((lane >> 4) << 3) + (j & 7) + ((j >> 3) << 4);
            int n = lane & 15;
            clfP[jj] = (n < NTAGS) ? (_Float16)clfw[(size_t)n * CLFK + k]
                                   : (_Float16)0.f;
        }
    }
}

// per-sequence reversed row map for the backward-direction input GEMM
__global__ void k_rev(const int* slen, int* rev) {
    int i = blockIdx.x * blockDim.x + threadIdx.x;
    if (i >= BB * SS) return;
    int b = i / SS, t = i % SS;
    int idx = slen[b] - 1 - t;
    if (idx < 0) idx = 0;
    rev[i] = b * SS + idx;
}

// ------------------------------------------------------------ embedding gather
__global__ void k_embed(const float* emb, const int* sentences, _Float16* x) {
    int word = blockIdx.x;                      // b*SS + s, 0..2047
    const float* row = emb + (size_t)sentences[word] * EDIM;
    for (int k = threadIdx.x; k < KPAD; k += blockDim.x) {
        if (k >= EDIM && k < DIN) continue;     // char features filled later
        _Float16 v = (_Float16)0.f;
        if (k < EDIM) v = (_Float16)row[k];
        x[(size_t)word * KPAD + k] = v;
    }
}

// ------------------------------------------- char BiLSTM: one wave per word/dir
__global__ void k_char_lstm(const float* cemb, const int* word_chars,
                            const int* word_len, const int* sent_len,
                            const float* wih_f, const float* whh_f, const float* b_f,
                            const float* wih_b, const float* whh_b, const float* b_b,
                            _Float16* x) {
    __shared__ float zbuf[8][200];
    __shared__ float hbuf[8][52];
    int wv = threadIdx.x >> 5, lane = threadIdx.x & 31;
    int gw = blockIdx.x * 8 + wv;               // 0..4095
    int word = gw >> 1, dir = gw & 1;
    int b = word / SS, s = word % SS;
    int len = word_len[word];
    const float* wih  = dir ? wih_b : wih_f;    // [200, 50]
    const float* whh  = dir ? whh_b : whh_f;    // [200, 50]
    const float* bias = dir ? b_b : b_f;

    for (int j = lane; j < CHH; j += 32) hbuf[wv][j] = 0.f;
    float c0 = 0.f, c1 = 0.f;                   // hidden j = lane, lane+32
    __syncthreads();

    for (int t = 0; t < TWC; ++t) {
        int tt = dir ? ((len - 1 - t) < 0 ? 0 : (len - 1 - t)) : t;
        const float* xrow = cemb + (size_t)word_chars[word * TWC + tt] * CEDIM;
#pragma unroll
        for (int jj = 0; jj < 7; ++jj) {
            int g = lane + 32 * jj;
            if (g < 200) {
                float z = bias[g];
                const float* wi = wih + (size_t)g * CEDIM;
                const float* wh = whh + (size_t)g * CEDIM;
                for (int k = 0; k < CEDIM; ++k)
                    z += xrow[k] * wi[k] + hbuf[wv][k] * wh[k];
                zbuf[wv][g] = z;
            }
        }
        __syncthreads();
        bool m = t < len;
#pragma unroll
        for (int h2 = 0; h2 < 2; ++h2) {
            int j = lane + 32 * h2;
            if (j < CHH) {
                float zi = zbuf[wv][j],         zf = zbuf[wv][CHH + j];
                float zg = zbuf[wv][2*CHH + j], zo = zbuf[wv][3*CHH + j];
                float cc = h2 ? c1 : c0;
                float cn = sigf(zf) * cc + sigf(zi) * tanh_(zg);
                float hn = sigf(zo) * tanh_(cn);
                if (m) { if (h2) c1 = cn; else c0 = cn; hbuf[wv][j] = hn; }
            }
        }
        __syncthreads();
    }
    bool sm = s < sent_len[b];
#pragma unroll
    for (int h2 = 0; h2 < 2; ++h2) {
        int j = lane + 32 * h2;
        if (j < CHH)
            x[(size_t)word * KPAD + EDIM + dir * CHH + j] =
                (_Float16)(sm ? hbuf[wv][j] : 0.f);
    }
}

// ------------------------- input-projection GEMM: gates = X @ WihT + b  (WMMA)
// M=2048 rows (optionally row-remapped for the reversed direction), N=1024, K=416
// Output stored TRANSPOSED as [t][g][b] so the recurrence can read each gate
// column for 8 consecutive batch rows with two b128 loads.
__global__ void k_gemm_gates(const _Float16* __restrict__ X,
                             const _Float16* __restrict__ WP,
                             const float* __restrict__ bias,
                             const int* row_map, float* __restrict__ out) {
    int lane = threadIdx.x & 31;
    int wid  = blockIdx.x * (blockDim.x >> 5) + (threadIdx.x >> 5);
    int mt = wid / NT_G;                        // 0..127
    int nt = wid % NT_G;                        // 0..63
    int row = mt * 16 + (lane & 15);
    if (row_map) row = row_map[row];
    const _Float16* arow = X + (size_t)row * KPAD + ((lane >> 4) << 3);

    v8f acc = {};
#pragma unroll
    for (int kt = 0; kt < KT_IH; ++kt) {
        if (kt + 1 < KT_IH)
            __builtin_prefetch((const v16h*)WP + (size_t)(nt * KT_IH + kt + 1) * 32, 0, 1);
        v16h a;
        const _Float16* p = arow + kt * 32;
#pragma unroll
        for (int j = 0; j < 8; ++j) { a[j] = p[j]; a[8 + j] = p[16 + j]; }
        v16h bf = load_b_packed(WP, nt * KT_IH + kt, lane);
        acc = wmma16(a, bf, acc);
    }
    int n = nt * 16 + (lane & 15);
    float bv = bias[n];
    int mbase = mt * 16 + ((lane >> 4) << 3);
#pragma unroll
    for (int r = 0; r < 8; ++r) {
        int rw = mbase + r;
        int b = rw >> 7;                        // / SS (tile never crosses b)
        int t = rw & (SS - 1);                  // % SS
        out[((size_t)t * G4 + n) * BB + b] = acc[r] + bv;
    }
}

__global__ void k_zero(float* p, long n) {
    for (long i = blockIdx.x * (long)blockDim.x + threadIdx.x; i < n;
         i += (long)gridDim.x * blockDim.x) p[i] = 0.f;
}

// -------------------- word BiLSTM recurrence: one persistent block per direction
// zh = h @ WhhT via WMMA (M=16 batch, K=256, N=1024); wave w owns n-tiles
// {w, w+16, w+32, w+48} => each lane holds matching i/f/g/o in registers.
__global__ void k_word_lstm(const float* __restrict__ gates,
                            const _Float16* __restrict__ whhP,
                            const int* sent_len, float* __restrict__ u) {
    __shared__ _Float16 hbuf[16][HID];          // 8 KB
    __shared__ int slen[16];
    int dir = blockIdx.x;
    const float* gx = gates + (size_t)dir * SS * G4 * BB;
    const _Float16* WP = whhP + (size_t)dir * HID * G4;
    int lane = threadIdx.x & 31, wv = threadIdx.x >> 5;   // 16 waves
    if (threadIdx.x < 16) slen[threadIdx.x] = sent_len[threadIdx.x];
    for (int i = threadIdx.x; i < 16 * HID; i += blockDim.x)
        (&hbuf[0][0])[i] = (_Float16)0.f;
    float creg[8];
#pragma unroll
    for (int r = 0; r < 8; ++r) creg[r] = 0.f;
    __syncthreads();

    int ncol = wv * 16 + (lane & 15);           // hidden column owned by lane
    int mhi  = (lane >> 4) * 8;

    for (int t = 0; t < SS; ++t) {
        v8f di = {}, df = {}, dg = {}, dq = {};
#pragma unroll
        for (int kt = 0; kt < KT_HH; ++kt) {
            const _Float16* ap = &hbuf[lane & 15][kt * 32 + ((lane >> 4) << 3)];
            v16h a;
#pragma unroll
            for (int j = 0; j < 8; ++j) { a[j] = ap[j]; a[8 + j] = ap[16 + j]; }
            v16h b0 = load_b_packed(WP, (wv     ) * KT_HH + kt, lane);
            v16h b1 = load_b_packed(WP, (wv + 16) * KT_HH + kt, lane);
            v16h b2 = load_b_packed(WP, (wv + 32) * KT_HH + kt, lane);
            v16h b3 = load_b_packed(WP, (wv + 48) * KT_HH + kt, lane);
            di = wmma16(a, b0, di);
            df = wmma16(a, b1, df);
            dg = wmma16(a, b2, dg);
            dq = wmma16(a, b3, dq);
        }
        // gate biases for this lane's column, 8 consecutive batch rows:
        // gates layout [t][g][b] -> two b128 loads per gate
        const float* gt = gx + (size_t)t * G4 * BB;
        float gi[8], gf[8], gg[8], go[8];
        *(float4*)&gi[0] = *(const float4*)(gt + (size_t)(0 * HID + ncol) * BB + mhi);
        *(float4*)&gi[4] = *(const float4*)(gt + (size_t)(0 * HID + ncol) * BB + mhi + 4);
        *(float4*)&gf[0] = *(const float4*)(gt + (size_t)(1 * HID + ncol) * BB + mhi);
        *(float4*)&gf[4] = *(const float4*)(gt + (size_t)(1 * HID + ncol) * BB + mhi + 4);
        *(float4*)&gg[0] = *(const float4*)(gt + (size_t)(2 * HID + ncol) * BB + mhi);
        *(float4*)&gg[4] = *(const float4*)(gt + (size_t)(2 * HID + ncol) * BB + mhi + 4);
        *(float4*)&go[0] = *(const float4*)(gt + (size_t)(3 * HID + ncol) * BB + mhi);
        *(float4*)&go[4] = *(const float4*)(gt + (size_t)(3 * HID + ncol) * BB + mhi + 4);
        __syncthreads();                        // all reads of hbuf done
#pragma unroll
        for (int r = 0; r < 8; ++r) {
            int m = mhi + r;                    // batch index
            float zi = di[r] + gi[r];
            float zf = df[r] + gf[r];
            float zg = dg[r] + gg[r];
            float zo = dq[r] + go[r];
            float cn = sigf(zf) * creg[r] + sigf(zi) * tanh_(zg);
            float hn = sigf(zo) * tanh_(cn);
            bool msk = t < slen[m];
            if (msk) { creg[r] = cn; hbuf[m][ncol] = (_Float16)hn; }
            if (dir == 0) {
                u[((size_t)t * BB + m) * H2 + ncol] = msk ? hn : 0.f;
            } else if (msk) {
                int tt = slen[m] - 1 - t;
                u[((size_t)tt * BB + m) * H2 + HID + ncol] = hn;
            }
        }
        __syncthreads();                        // hbuf updated for next step
    }
}

// ---------------------------------------------------------------- prefix sums
__global__ void k_csum(const float* __restrict__ u, float* __restrict__ csum) {
    int i = blockIdx.x * blockDim.x + threadIdx.x;  // b*H2 + f
    if (i >= BB * H2) return;
    int b = i / H2, f = i % H2;
    float run = 0.f;
    csum[(size_t)b * H2 + f] = 0.f;
    for (int t = 0; t < SS; ++t) {
        run += u[((size_t)t * BB + b) * H2 + f];
        csum[((size_t)(t + 1) * BB + b) * H2 + f] = run;
    }
}

// ------------- classifier GEMM: one wave per window; A = relu(first|mean|last)
__global__ void k_classifier(const float* __restrict__ u,
                             const float* __restrict__ csum,
                             const _Float16* __restrict__ clfP,
                             const float* __restrict__ clfb,
                             float* __restrict__ out) {
    int lane = threadIdx.x & 31;
    int win  = blockIdx.x * (blockDim.x >> 5) + (threadIdx.x >> 5);
    if (win >= NWIN) return;
    // decode (size r, start) from the r=1..16 concatenated window list
    int r = 1, acc0 = 0;
    while (win >= acc0 + (SS - r + 1)) { acc0 += SS - r + 1; ++r; }
    int start = win - acc0;
    float inv = 1.f / (float)r;

    int m  = lane & 15;                         // batch index
    int ko = (lane >> 4) << 3;
    v8f acc = {};
#pragma unroll 4
    for (int kt = 0; kt < KT_CL; ++kt) {
        int seg = kt >> 4;                      // 0=first, 1=mean, 2=last
        int klo = (kt & 15) * 32;
        const float* p1; const float* p2 = nullptr; float scl = 1.f;
        if (seg == 0)      p1 = u + ((size_t)start * BB + m) * H2;
        else if (seg == 2) p1 = u + ((size_t)(start + r - 1) * BB + m) * H2;
        else { p1 = csum + ((size_t)(start + r) * BB + m) * H2;
               p2 = csum + ((size_t)start * BB + m) * H2; scl = inv; }
        v16h a;
#pragma unroll
        for (int j = 0; j < 8; ++j) {
            int f0 = klo + ko + j, f1 = klo + ko + 16 + j;
            float v0 = p2 ? (p1[f0] - p2[f0]) * scl : p1[f0];
            float v1 = p2 ? (p1[f1] - p2[f1]) * scl : p1[f1];
            a[j]     = (_Float16)fmaxf(v0, 0.f);
            a[8 + j] = (_Float16)fmaxf(v1, 0.f);
        }
        v16h bf = load_b_packed(clfP, kt, lane);
        acc = wmma16(a, bf, acc);
    }
    int tag = lane & 15;
    if (tag < NTAGS) {
        float bv = clfb[tag];
        int mhi = (lane >> 4) * 8;
#pragma unroll
        for (int rr = 0; rr < 8; ++rr) {
            int b = mhi + rr;
            out[((size_t)b * NTAGS + tag) * NWIN + win] = acc[rr] + bv;
        }
    }
}

// ------------------------------------------------------------------- launcher
extern "C" void kernel_launch(void* const* d_in, const int* in_sizes, int n_in,
                              void* d_out, int out_size, void* d_ws, size_t ws_size,
                              hipStream_t stream) {
    const float* emb      = (const float*)d_in[0];
    const float* char_emb = (const float*)d_in[1];
    const float* c_wih_f  = (const float*)d_in[2];
    const float* c_whh_f  = (const float*)d_in[3];
    const float* c_b_f    = (const float*)d_in[4];
    const float* c_wih_b  = (const float*)d_in[5];
    const float* c_whh_b  = (const float*)d_in[6];
    const float* c_b_b    = (const float*)d_in[7];
    const float* w_wih_f  = (const float*)d_in[8];
    const float* w_whh_f  = (const float*)d_in[9];
    const float* w_b_f    = (const float*)d_in[10];
    const float* w_wih_b  = (const float*)d_in[11];
    const float* w_whh_b  = (const float*)d_in[12];
    const float* w_b_b    = (const float*)d_in[13];
    const float* clf_w    = (const float*)d_in[14];
    const float* clf_b    = (const float*)d_in[15];
    const int* sentences  = (const int*)d_in[16];
    const int* sent_len   = (const int*)d_in[17];
    const int* word_chars = (const int*)d_in[18];
    const int* word_len   = (const int*)d_in[19];
    float* out = (float*)d_out;

    char* ws = (char*)d_ws;
    auto take = [&](size_t bytes) {
        char* p = ws; ws += (bytes + 255) & ~(size_t)255; return p;
    };
    _Float16* x16  = (_Float16*)take((size_t)2048 * KPAD * 2);
    _Float16* wihP = (_Float16*)take((size_t)2 * KPAD * G4 * 2);
    _Float16* whhP = (_Float16*)take((size_t)2 * HID * G4 * 2);
    _Float16* clfP = (_Float16*)take((size_t)CLFK * 16 * 2);
    float* gates   = (float*)take((size_t)2 * SS * G4 * BB * 4);
    float* u       = (float*)take((size_t)SS * BB * H2 * 4);
    float* csum    = (float*)take((size_t)(SS + 1) * BB * H2 * 4);
    int*   rev     = (int*)take((size_t)2048 * 4);

    k_pack_weights<<<256, 256, 0, stream>>>(w_wih_f, w_wih_b, w_whh_f, w_whh_b,
                                            clf_w, wihP, whhP, clfP);
    k_rev<<<(BB * SS + 255) / 256, 256, 0, stream>>>(sent_len, rev);
    k_embed<<<2048, 128, 0, stream>>>(emb, sentences, x16);
    k_char_lstm<<<512, 256, 0, stream>>>(char_emb, word_chars, word_len, sent_len,
                                         c_wih_f, c_whh_f, c_b_f,
                                         c_wih_b, c_whh_b, c_b_b, x16);
    k_gemm_gates<<<1024, 256, 0, stream>>>(x16, wihP, w_b_f, nullptr, gates);
    k_gemm_gates<<<1024, 256, 0, stream>>>(x16, wihP + (size_t)KPAD * G4, w_b_b,
                                           rev, gates + (size_t)SS * G4 * BB);
    k_zero<<<1024, 256, 0, stream>>>(u, (long)SS * BB * H2);
    k_word_lstm<<<2, 512, 0, stream>>>(gates, whhP, sent_len, u);
    k_csum<<<(BB * H2 + 255) / 256, 256, 0, stream>>>(u, csum);
    k_classifier<<<(NWIN + 3) / 4, 128, 0, stream>>>(u, csum, clfP, clf_b, out);
}
